// DeformConv_755914244225
// MI455X (gfx1250) — compile-verified, matrix operand order checked by reading, not run on hardware
//
#include <hip/hip_runtime.h>

typedef __attribute__((ext_vector_type(16))) _Float16 v16h;
typedef __attribute__((ext_vector_type(8)))  _Float16 v8h;
typedef __attribute__((ext_vector_type(8)))  float    v8f;

#define B_ 8
#define C_ 128
#define H_ 64
#define W_ 64
#define O_ 128
#define KK 1152   // C*9

// ---------------------------------------------------------------------------
// Kernel 1: pack conv_w (O,C,3,3) f32 -> f16, K reordered as k = tap*C + c
// so each WMMA K-step of 32 stays inside a single tap (gather params shared
// across all 32 channels of a B fragment).
// ---------------------------------------------------------------------------
__global__ void pack_w_kernel(const float* __restrict__ conv_w,
                              _Float16* __restrict__ w16) {
    int idx = blockIdx.x * blockDim.x + threadIdx.x;
    if (idx >= O_ * KK) return;
    int o = idx / KK;
    int r = idx - o * KK;
    int t = r >> 7;       // tap
    int c = r & 127;      // channel
    w16[idx] = (_Float16)conv_w[o * KK + c * 9 + t];
}

// ---------------------------------------------------------------------------
// Kernel 2: offset(18ch) + mask(9ch, sigmoid) 3x3 conv.
// One thread per output pixel; weights staged through LDS in 16-ch chunks.
// Outputs dy/dx/mask as (B,9,H,W) f32 arrays in workspace.
// ---------------------------------------------------------------------------
__global__ void __launch_bounds__(256) offmask_kernel(
    const float* __restrict__ x,
    const float* __restrict__ off_w,  const float* __restrict__ off_b,
    const float* __restrict__ mask_w, const float* __restrict__ mask_b,
    float* __restrict__ dyA, float* __restrict__ dxA, float* __restrict__ mA)
{
    __shared__ float wbuf[27 * 16 * 9];   // [oc][ci][tap], 15.2 KB
    int pix = blockIdx.x * 256 + threadIdx.x;
    int b = pix >> 12, h = (pix >> 6) & 63, w = pix & 63;
    const float* xb = x + ((size_t)b << 19);   // b * C*H*W

    float acc[27];
    #pragma unroll
    for (int i = 0; i < 27; ++i) acc[i] = 0.f;

    for (int c0 = 0; c0 < C_; c0 += 16) {
        __syncthreads();
        for (int e = threadIdx.x; e < 27 * 16 * 9; e += 256) {
            int oc  = e / 144;
            int rem = e - oc * 144;
            int ci  = rem / 9;
            int t   = rem - ci * 9;
            int c   = c0 + ci;
            wbuf[e] = (oc < 18) ? off_w[oc * KK + c * 9 + t]
                                : mask_w[(oc - 18) * KK + c * 9 + t];
        }
        __syncthreads();
        for (int ci = 0; ci < 16; ++ci) {
            const float* xc = xb + ((size_t)(c0 + ci) << 12);
            float xv[9];
            #pragma unroll
            for (int t = 0; t < 9; ++t) {
                int yy = h - 1 + t / 3;
                int xx = w - 1 + t % 3;
                bool ok = (yy >= 0) & (yy < H_) & (xx >= 0) & (xx < W_);
                xv[t] = ok ? xc[(yy << 6) + xx] : 0.f;
            }
            const float* wr = &wbuf[ci * 9];
            #pragma unroll
            for (int oc = 0; oc < 27; ++oc) {
                #pragma unroll
                for (int t = 0; t < 9; ++t)
                    acc[oc] += xv[t] * wr[oc * 144 + t];
            }
        }
    }

    #pragma unroll
    for (int t = 0; t < 9; ++t) {
        float dy = acc[2 * t]     + off_b[2 * t];
        float dx = acc[2 * t + 1] + off_b[2 * t + 1];
        float z  = acc[18 + t]    + mask_b[t];
        float m  = 1.f / (1.f + __expf(-z));
        int oi = (((b * 9 + t) << 6) + h) * 64 + w;
        dyA[oi] = dy; dxA[oi] = dx; mA[oi] = m;
    }
}

// ---------------------------------------------------------------------------
// Kernel 3: fused gather + mask + GEMM with software-pipelined, double-
// buffered B tiles. Block = 256 threads = 8 waves, tile 128(O) x 64(pixels)
// (one full image row -> b,h block-uniform). Wave tile: 16(O) x 64(N) = 4
// accumulators sharing one A fragment. 36 K-stages (9 taps x 4 c-chunks).
// All four B fragments preloaded into distinct registers so the 4 WMMAs
// issue back-to-back behind a single s_wait_dscnt.
// ---------------------------------------------------------------------------
__device__ __forceinline__ void build_tile(
    int s, _Float16* __restrict__ buf,
    const float* __restrict__ x,
    const float* __restrict__ dyA, const float* __restrict__ dxA,
    const float* __restrict__ mA,
    int b, int h, int tid)
{
    const int t  = s >> 2;       // tap 0..8
    const int cc = s & 3;        // channel chunk
    const int ty = t / 3, tx = t % 3;
    #pragma unroll
    for (int i = 0; i < 8; ++i) {
        int v  = tid + (i << 8);
        int p  = v >> 5;         // pixel 0..63 (wave-uniform)
        int cl = v & 31;         // channel-in-chunk = K index in fragment
        int w  = p;              // p0 is row-aligned -> w == p
        int oi = (((b * 9 + t) << 6) + h) * 64 + w;
        float py = (float)(h - 1 + ty) + dyA[oi];
        float px = (float)(w - 1 + tx) + dxA[oi];
        float m  = mA[oi];
        float fy = floorf(py), fx = floorf(px);
        int   y0 = (int)fy,    x0 = (int)fx;
        float wy = py - fy,    wx = px - fx;
        int c = (cc << 5) + cl;
        const float* bp = x + (((size_t)(b << 7) + c) << 12);
        bool yv0 = (y0 >= 0) & (y0 < H_);
        bool yv1 = (y0 + 1 >= 0) & (y0 + 1 < H_);
        bool xv0 = (x0 >= 0) & (x0 < W_);
        bool xv1 = (x0 + 1 >= 0) & (x0 + 1 < W_);
        float v00 = (yv0 & xv0) ? bp[(y0 << 6) + x0]           : 0.f;
        float v01 = (yv0 & xv1) ? bp[(y0 << 6) + x0 + 1]       : 0.f;
        float v10 = (yv1 & xv0) ? bp[((y0 + 1) << 6) + x0]     : 0.f;
        float v11 = (yv1 & xv1) ? bp[((y0 + 1) << 6) + x0 + 1] : 0.f;
        float bi = (v00 * (1.f - wx) + v01 * wx) * (1.f - wy)
                 + (v10 * (1.f - wx) + v11 * wx) * wy;
        buf[(p << 5) + cl] = (_Float16)(bi * m);
    }
}

__global__ void __launch_bounds__(256) dcn_main_kernel(
    const float*    __restrict__ x,
    const _Float16* __restrict__ w16,
    const float* __restrict__ dyA, const float* __restrict__ dxA,
    const float* __restrict__ mA,
    float* __restrict__ out)
{
    __shared__ __align__(16) _Float16 Bt[2][64 * 32];  // double-buffered [pixel][c]

    const int tid  = threadIdx.x;
    const int lane = tid & 31;       // wave32
    const int wave = tid >> 5;       // 0..7
    const int o0   = wave << 4;      // 16 output channels per wave
    const int b    = blockIdx.x >> 6;        // p0 = blockIdx*64 -> full row
    const int h    = blockIdx.x & 63;

    const int r_  = lane & 15;       // A row / B col / D col
    const int hi_ = lane >> 4;       // lane-half selector per ISA layouts

    v8f acc[4] = {};

    // Prologue: build stage 0.
    build_tile(0, Bt[0], x, dyA, dxA, mA, b, h, tid);

    for (int s = 0; s < 36; ++s) {
        __syncthreads();
        // Pipeline: issue next stage's gathers while this stage's WMMAs run.
        if (s + 1 < 36)
            build_tile(s + 1, Bt[(s + 1) & 1], x, dyA, dxA, mA, b, h, tid);

        const _Float16* bb = Bt[s & 1];
        const int t  = s >> 2;
        const int cc = s & 3;
        const int k0 = (t << 7) + (cc << 5);

        // A fragment: 16x32 f16 weights from L2-resident packed buffer.
        // ISA A layout: lane l (hi=l>>4): halves 0..7 = K(hi*8..hi*8+7),
        // halves 8..15 = K(16+hi*8..23+hi*8) -> two contiguous b128 loads.
        v16h af;
        {
            const _Float16* ar = w16 + (size_t)(o0 + r_) * KK + k0 + (hi_ << 3);
            ((v8h*)&af)[0] = *(const v8h*)(ar);
            ((v8h*)&af)[1] = *(const v8h*)(ar + 16);
        }

        // Preload all 4 B fragments (distinct registers) so the ds_loads
        // clause together and the 4 WMMAs issue back-to-back.
        v16h bf[4];
        #pragma unroll
        for (int j = 0; j < 4; ++j) {
            const _Float16* br = bb + ((((j << 4) + r_) << 5) + (hi_ << 3));
            ((v8h*)&bf[j])[0] = *(const v8h*)(br);
            ((v8h*)&bf[j])[1] = *(const v8h*)(br + 16);
        }
        #pragma unroll
        for (int j = 0; j < 4; ++j) {
            acc[j] = __builtin_amdgcn_wmma_f32_16x16x32_f16(
                         false, af, false, bf[j], (short)0, acc[j], false, false);
        }
    }

    // D epilogue: C/D layout -> VGPR r, lane: M = r + 8*hi_, N = r_.
    #pragma unroll
    for (int j = 0; j < 4; ++j) {
        int w = (j << 4) + r_;
        #pragma unroll
        for (int r = 0; r < 8; ++r) {
            int o = o0 + r + (hi_ << 3);
            out[(((size_t)(b << 7) + o) << 12) + (h << 6) + w] = acc[j][r];
        }
    }
}

// ---------------------------------------------------------------------------
extern "C" void kernel_launch(void* const* d_in, const int* in_sizes, int n_in,
                              void* d_out, int out_size, void* d_ws, size_t ws_size,
                              hipStream_t stream)
{
    const float* x      = (const float*)d_in[0];
    const float* conv_w = (const float*)d_in[1];
    const float* off_w  = (const float*)d_in[2];
    const float* off_b  = (const float*)d_in[3];
    const float* mask_w = (const float*)d_in[4];
    const float* mask_b = (const float*)d_in[5];
    float* out = (float*)d_out;

    char* ws = (char*)d_ws;
    _Float16* w16 = (_Float16*)ws;                        // 128*1152*2 = 294912 B
    float* dyA = (float*)(ws + 294912);                   // B*9*H*W*4 = 1179648 B
    float* dxA = (float*)(ws + 294912 + 1179648);
    float* mA  = (float*)(ws + 294912 + 2 * 1179648);

    pack_w_kernel<<<(O_ * KK + 255) / 256, 256, 0, stream>>>(conv_w, w16);
    offmask_kernel<<<(B_ * H_ * W_) / 256, 256, 0, stream>>>(
        x, off_w, off_b, mask_w, mask_b, dyA, dxA, mA);
    dcn_main_kernel<<<(B_ * H_ * W_) / 64, 256, 0, stream>>>(
        x, w16, dyA, dxA, mA, out);
}
